// LSTMCell_14903536518016
// MI455X (gfx1250) — matrix operand order repaired and checked by reading, not compiled, not run on hardware
//
#include <hip/hip_runtime.h>

// Problem constants (match reference)
constexpr int Bsz = 8192;
constexpr int Dsz = 2048;
constexpr int Hsz = 2048;
constexpr int NG  = 7;
constexpr int NGH = NG * Hsz;     // 14336

// Tiling
constexpr int BM = 128;
constexpr int BN = 32;
constexpr int BK = 32;
constexpr int LDA = BK + 8;       // padded LDS row (bf16 elems)
constexpr int LDB = BK + 8;
constexpr int ASZ = BM * LDA;     // 5120 ushorts per A buffer
constexpr int WSZ = NG * BN * LDB;// 8960 ushorts per W buffer
constexpr int KT  = Dsz / BK;     // 64 K-tiles

typedef __attribute__((ext_vector_type(16))) __bf16 v16bf;
typedef __attribute__((ext_vector_type(8)))  __bf16 v8bf;
typedef __attribute__((ext_vector_type(8)))  float  v8f;

// ---- fp32 -> bf16 (RNE), packed where hardware supports it ----
#if __has_builtin(__builtin_amdgcn_cvt_pk_bf16_f32)
typedef __attribute__((ext_vector_type(2))) __bf16 v2bf;
static __device__ inline unsigned pk2bf(float lo, float hi) {
    v2bf p = __builtin_amdgcn_cvt_pk_bf16_f32(lo, hi);
    return __builtin_bit_cast(unsigned, p);
}
#else
static __device__ inline unsigned short f2bf_(float f) {
    unsigned u = __float_as_uint(f);
    u += 0x7FFFu + ((u >> 16) & 1u);
    return (unsigned short)(u >> 16);
}
static __device__ inline unsigned pk2bf(float lo, float hi) {
    return (unsigned)f2bf_(lo) | ((unsigned)f2bf_(hi) << 16);
}
#endif
static __device__ inline unsigned short one_bf(float x) { return (unsigned short)pk2bf(x, x); }

static __device__ inline float sigmoidf_(float x) { return 1.0f / (1.0f + __expf(-x)); }
static __device__ inline float tanhf_(float x)    { return 1.0f - 2.0f / (__expf(2.0f * x) + 1.0f); }
static __device__ inline float softplusf_(float x){ return (x > 15.0f) ? x : __logf(1.0f + __expf(x)); }

// Build a 16-element bf16 fragment from two contiguous 16-byte LDS chunks.
static __device__ inline v16bf ldfrag(const unsigned short* p0, const unsigned short* p1) {
    v8bf lo = *(const v8bf*)p0;
    v8bf hi = *(const v8bf*)p1;
    return __builtin_shufflevector(lo, hi, 0,1,2,3,4,5,6,7,8,9,10,11,12,13,14,15);
}

// ---- CDNA5 async global->LDS staging (ASYNCcnt path) ----
#if __has_builtin(__builtin_amdgcn_global_load_async_to_lds_b128)
#define HAVE_ASYNC_LDS 1
typedef int v4i __attribute__((vector_size(16)));
static __device__ inline void async_cp16(const unsigned short* g, unsigned short* l) {
    // per-lane 16-byte transfer, memory -> LDS, no VGPR data round-trip
    __builtin_amdgcn_global_load_async_to_lds_b128((v4i*)g, (v4i*)l, 0, 0);
}
static __device__ inline void wait_async0() {
#if __has_builtin(__builtin_amdgcn_s_wait_asynccnt)
    __builtin_amdgcn_s_wait_asynccnt(0);
#else
    asm volatile("s_wait_asynccnt 0" ::: "memory");
#endif
}
#endif

// ================= prep kernels (workspace path) =================

// hx fp32 [B,D] -> bf16 [B,D]
__global__ __launch_bounds__(256) void conv_hx(const float* __restrict__ hx,
                                               unsigned short* __restrict__ hxb) {
    const size_t i = ((size_t)blockIdx.x * blockDim.x + threadIdx.x) * 4;
    const float4 v = *(const float4*)(hx + i);
    *(uint2*)(hxb + i) = make_uint2(pk2bf(v.x, v.y), pk2bf(v.z, v.w));
}

// W fp32 [D, 7H] (n contiguous) -> bf16 Wt [g][n][k] (k contiguous)
__global__ __launch_bounds__(256) void conv_wt(const float* __restrict__ Wm,
                                               unsigned short* __restrict__ Wtb) {
    __shared__ unsigned short T[32 * 33];
    const int tid  = threadIdx.x;
    const int kBlk = blockIdx.x * 32;   // over D
    const int nBlk = blockIdx.y * 32;   // over 7H (tiles never straddle a gate: H%32==0)
    #pragma unroll
    for (int i = 0; i < 4; ++i) {
        const int e = tid + 256 * i;          // 0..1023
        const int r = e >> 5, c = e & 31;     // k-row, n-col
        T[c * 33 + r] = one_bf(Wm[(size_t)(kBlk + r) * NGH + nBlk + c]);
    }
    __syncthreads();
    #pragma unroll
    for (int i = 0; i < 4; ++i) {
        const int e = tid + 256 * i;
        const int c = e >> 5, r = e & 31;     // write k-contiguous
        const int nGlob = nBlk + c;
        const int g = nGlob >> 11;            // / Hsz
        const int n = nGlob & (Hsz - 1);
        Wtb[(size_t)g * Hsz * Dsz + (size_t)n * Dsz + kBlk + r] = T[c * 33 + r];
    }
}

// ================= main fused GEMM + cell kernel =================
// MODE 0: convert fp32 -> bf16 while staging (no workspace needed)
// MODE 1: stage pre-converted bf16 (hxb, Wtb); async LDS path if available

template <int MODE>
__global__ __launch_bounds__(256, 1) void clstm_wmma(
    const float* __restrict__ hx,            // [B, D] fp32
    const float* __restrict__ cx1,           // [B, H]
    const float* __restrict__ cx2,           // [B, H]
    const float* __restrict__ dt,            // [B, 1]
    const float* __restrict__ Wm,            // [D, 7H] fp32
    const float* __restrict__ bv,            // [7H]
    const unsigned short* __restrict__ hxb,  // [B, D] bf16       (MODE 1)
    const unsigned short* __restrict__ Wtb,  // [g][n][k] bf16    (MODE 1)
    float* __restrict__ out)                 // [3, B, H]
{
    __shared__ unsigned short As[2 * ASZ];   // ping-pong A tiles (row-major bf16)
    __shared__ unsigned short Ws[2 * WSZ];   // ping-pong W tiles ([g][n][k] bf16)

    const int tid   = threadIdx.x;
    const int lane  = tid & 31;
    const int wid   = tid >> 5;
    const int waveM = wid & 3;        // 4 wave-rows of 32
    const int waveN = wid >> 2;       // 2 wave-cols of 16
    const int mBlock = blockIdx.x * BM;
    const int nBlock = blockIdx.y * BN;

    const int half  = lane >> 4;               // K-half selector per WMMA layout
    const int rowA0 = waveM * 32 + (lane & 15);
    const int colB  = waveN * 16 + (lane & 15);

    v8f acc[NG][2];
    {
        v8f z = {0.f,0.f,0.f,0.f,0.f,0.f,0.f,0.f};
        #pragma unroll
        for (int g = 0; g < NG; ++g) { acc[g][0] = z; acc[g][1] = z; }
    }

    auto compute = [&](int bufi) {
        const unsigned short* A  = As + bufi * ASZ;
        const unsigned short* Wt = Ws + bufi * WSZ;
        const v16bf a0 = ldfrag(&A[(rowA0     ) * LDA + half * 8],
                                &A[(rowA0     ) * LDA + 16 + half * 8]);
        const v16bf a1 = ldfrag(&A[(rowA0 + 16) * LDA + half * 8],
                                &A[(rowA0 + 16) * LDA + 16 + half * 8]);
        #pragma unroll
        for (int g = 0; g < NG; ++g) {
            const unsigned short* pb = &Wt[g * (BN * LDB) + colB * LDB + half * 8];
            const v16bf bfr = ldfrag(pb, pb + 16);
            acc[g][0] = __builtin_amdgcn_wmma_f32_16x16x32_bf16(
                false, a0, false, bfr, (short)0, acc[g][0], false, false);
            acc[g][1] = __builtin_amdgcn_wmma_f32_16x16x32_bf16(
                false, a1, false, bfr, (short)0, acc[g][1], false, false);
        }
    };

#if defined(HAVE_ASYNC_LDS)
    if constexpr (MODE == 1) {
        // ---- async staging: 6 x 16B per lane per tile, ASYNCcnt-tracked ----
        // A tile: 512 chunks of 16B -> 2 per thread
        size_t  aoff[2]; unsigned aldb[2];
        #pragma unroll
        for (int i = 0; i < 2; ++i) {
            const int c = tid + 256 * i;
            const int r = c >> 2, sb = c & 3;
            aoff[i] = (size_t)(mBlock + r) * Dsz + sb * 8;
            aldb[i] = r * LDA + sb * 8;
        }
        // W tile: 896 chunks of 16B -> 4 per thread (mod wrap: surplus lanes
        // re-copy identical data; benign, keeps per-wave ASYNCcnt uniform)
        size_t  woff[4]; unsigned wldb[4];
        #pragma unroll
        for (int j = 0; j < 4; ++j) {
            const int c = (tid + 256 * j) % (NG * 32 * 4);
            const int r = c >> 2, sb = c & 3;
            const int g = r >> 5, n = r & 31;
            woff[j] = ((size_t)g * Hsz + nBlock + n) * Dsz + sb * 8;
            wldb[j] = g * (BN * LDB) + n * LDB + sb * 8;
        }
        auto stage_async = [&](int k0, int bufi) {
            unsigned short* Ab = As + bufi * ASZ;
            unsigned short* Wb = Ws + bufi * WSZ;
            #pragma unroll
            for (int i = 0; i < 2; ++i) async_cp16(hxb + aoff[i] + k0, Ab + aldb[i]);
            #pragma unroll
            for (int j = 0; j < 4; ++j) async_cp16(Wtb + woff[j] + k0, Wb + wldb[j]);
        };

        stage_async(0, 0);
        #pragma unroll 1
        for (int t = 0; t < KT; ++t) {
            wait_async0();                                  // tile t landed (this wave)
            __syncthreads();                                // all waves landed; prev reads done
            if (t + 1 < KT) stage_async((t + 1) * BK, (t + 1) & 1);  // overlaps compute
            compute(t & 1);
        }
    } else
#endif
    {
        // ---- synchronous staging (short live ranges: load right before store) ----
        auto prefetch_tile = [&](int k0) {
            if constexpr (MODE == 0) {
                __builtin_prefetch(hx + (size_t)(mBlock + (tid >> 3)) * Dsz + k0, 0, 0);
                __builtin_prefetch(Wm + (size_t)(k0 + (tid >> 3)) * NGH + nBlock, 0, 0);
            } else {
                __builtin_prefetch(hxb + (size_t)(mBlock + (tid >> 1)) * Dsz + k0, 0, 0);
                __builtin_prefetch(Wtb + ((size_t)(tid >> 5) * Hsz + nBlock + (tid & 31)) * Dsz + k0, 0, 0);
            }
        };

        auto stage_tile = [&](int k0, int bufi) {
            unsigned short* A  = As + bufi * ASZ;
            unsigned short* Wt = Ws + bufi * WSZ;
            if constexpr (MODE == 0) {
                #pragma unroll
                for (int i = 0; i < 4; ++i) {
                    const int idx = tid + 256 * i;
                    const float4 v = *(const float4*)(hx + (size_t)(mBlock + (idx >> 3)) * Dsz
                                                         + k0 + (idx & 7) * 4);
                    *(uint2*)&A[(idx >> 3) * LDA + (idx & 7) * 4] =
                        make_uint2(pk2bf(v.x, v.y), pk2bf(v.z, v.w));
                }
                const int k = tid >> 3, nq = tid & 7;
                #pragma unroll
                for (int g = 0; g < NG; ++g) {
                    const float4 v = *(const float4*)(Wm + (size_t)(k0 + k) * NGH
                                                         + (size_t)g * Hsz + nBlock + nq * 4);
                    const int base = g * (BN * LDB) + (nq * 4) * LDB + k;
                    Wt[base          ] = one_bf(v.x);
                    Wt[base +     LDB] = one_bf(v.y);
                    Wt[base + 2 * LDB] = one_bf(v.z);
                    Wt[base + 3 * LDB] = one_bf(v.w);
                }
            } else {
                const int arow = tid >> 1, ac = tid & 1;
                const unsigned short* pa = hxb + (size_t)(mBlock + arow) * Dsz + k0 + ac * 16;
                unsigned short* da = &A[arow * LDA + ac * 16];
                *(uint4*)(da)     = *(const uint4*)(pa);
                *(uint4*)(da + 8) = *(const uint4*)(pa + 8);
                if (tid < NG * 32) {
                    const int wg = tid >> 5, wn = tid & 31;
                    const unsigned short* pw = Wtb + ((size_t)wg * Hsz + nBlock + wn) * Dsz + k0;
                    unsigned short* dw = &Wt[wg * (BN * LDB) + wn * LDB];
                    #pragma unroll
                    for (int i = 0; i < 4; ++i) *(uint4*)(dw + 8 * i) = *(const uint4*)(pw + 8 * i);
                }
            }
        };

        stage_tile(0, 0);
        #pragma unroll 1
        for (int t = 0; t < KT; ++t) {
            __syncthreads();                        // buf[t&1] staged by all waves
            if (t + 1 < KT) prefetch_tile((t + 1) * BK);
            compute(t & 1);
            if (t + 1 < KT) stage_tile((t + 1) * BK, (t + 1) & 1);
        }
    }

    // ---- fused epilogue: bias + gate nonlinearity + CT-decay cell update ----
    const int col = nBlock + colB;
    float bias[NG];
    #pragma unroll
    for (int g = 0; g < NG; ++g) bias[g] = bv[(size_t)g * Hsz + col];

    const size_t plane = (size_t)Bsz * Hsz;
    #pragma unroll
    for (int s = 0; s < 2; ++s) {
        const int tileRow = mBlock + waveM * 32 + s * 16 + half * 8;
        #pragma unroll
        for (int r = 0; r < 8; ++r) {
            const int row = tileRow + r;
            const float i1 = sigmoidf_(acc[0][s][r] + bias[0]);
            const float i2 = sigmoidf_(acc[1][s][r] + bias[1]);
            const float f1 = sigmoidf_(acc[2][s][r] + bias[2]);
            const float f2 = sigmoidf_(acc[3][s][r] + bias[3]);
            const float o  = sigmoidf_(acc[4][s][r] + bias[4]);
            const float z  = tanhf_   (acc[5][s][r] + bias[5]);
            const float dc = softplusf_(acc[6][s][r] + bias[6]);

            const size_t ix = (size_t)row * Hsz + col;
            const float c1 = cx1[ix];
            const float c2 = cx2[ix];
            const float cy1 = f1 * c1 + i1 * z;
            const float cy2 = f2 * c2 + i2 * z;
            const float e   = __expf(-dc * dt[row]);   // t - tj == dt
            const float ct  = cy2 + (cy1 - cy2) * e;
            const float h   = o * tanhf_(ct);

            out[ix]             = cy1;
            out[plane + ix]     = cy2;
            out[2 * plane + ix] = h;
        }
    }
}

extern "C" void kernel_launch(void* const* d_in, const int* in_sizes, int n_in,
                              void* d_out, int out_size, void* d_ws, size_t ws_size,
                              hipStream_t stream) {
    (void)in_sizes; (void)n_in; (void)out_size;
    const float* hx  = (const float*)d_in[0];
    const float* cx1 = (const float*)d_in[1];
    const float* cx2 = (const float*)d_in[2];
    /* d_in[3] = tj, unused: t - tj == dt */
    const float* dt  = (const float*)d_in[4];
    const float* Wm  = (const float*)d_in[5];
    const float* bv  = (const float*)d_in[6];
    float* out = (float*)d_out;

    const size_t hxbBytes = (size_t)Bsz * Dsz * sizeof(unsigned short);      // 32 MiB
    const size_t wtbBytes = (size_t)NG * Hsz * Dsz * sizeof(unsigned short); // 56 MiB
    const size_t need = hxbBytes + wtbBytes;

    dim3 grid(Bsz / BM, Hsz / BN);   // 64 x 64
    dim3 block(256);                 // 8 wave32

    if (ws_size >= need) {
        unsigned short* hxb = (unsigned short*)d_ws;
        unsigned short* Wtb = (unsigned short*)((char*)d_ws + hxbBytes);
        // one-time (per launch) fp32->bf16 conversion + W transpose to [g][n][k]
        conv_hx<<<dim3((Bsz * Dsz) / (256 * 4)), block, 0, stream>>>(hx, hxb);
        conv_wt<<<dim3(Dsz / 32, NGH / 32), block, 0, stream>>>(Wm, Wtb);
        clstm_wmma<1><<<grid, block, 0, stream>>>(hx, cx1, cx2, dt, Wm, bv, hxb, Wtb, out);
    } else {
        clstm_wmma<0><<<grid, block, 0, stream>>>(hx, cx1, cx2, dt, Wm, bv, nullptr, nullptr, out);
    }
}
